// gtnet_65730179498155
// MI455X (gfx1250) — compile-verified, matrix operand order checked by reading, not run on hardware
//
#include <hip/hip_runtime.h>
#include <cstddef>
#include <cstdint>

// ---------------------------------------------------------------------------
// MTGNN/DSTAGNN gtnet forward for MI455X (gfx1250, wave32).
//
// Heavy math = batched 500x500x500 spatial-attention GEMM and Chebyshev
// 500x500x(T*32) GEMMs (~35-40 GFLOP/layer); operands fit in the 192 MB L2 so
// they are matrix-pipe bound -> v_wmma_f32_16x16x32_bf16 with fp32 accumulate.
// Both GEMM operands are packed into bf16 staging buffers with K padded to a
// multiple of 32 (Kp=512) and B stored transposed, so every WMMA fragment is
// exactly two contiguous 16-byte loads per lane: the inner loop is
// 8x global_load_b128 + 4x v_wmma per K-step, with zero divergence.
// Everything else is bandwidth-trivial VALU work.
// ---------------------------------------------------------------------------

typedef __attribute__((ext_vector_type(16))) __bf16 v16bf;
typedef __attribute__((ext_vector_type(8)))  __bf16 v8bf;
typedef __attribute__((ext_vector_type(8)))  float  v8f;

static constexpr int B_  = 32;   // batch
static constexpr int N_  = 500;  // graph nodes
static constexpr int F_  = 32;   // feature channels
static constexpr int KP_ = 512;  // K padded to multiple of 32 (K=500)

#define DEV __device__ __forceinline__
DEV float sigm_(float x) { return 1.0f / (1.0f + expf(-x)); }
DEV float relu_(float x) { return x > 0.0f ? x : 0.0f; }
DEV long gtid_() { return (long)blockIdx.x * blockDim.x + threadIdx.x; }

// ------------------------------ tiny utils ---------------------------------
__global__ void k_zero(float* p, long total) {
    long i = gtid_(); if (i < total) p[i] = 0.0f;
}
__global__ void k_copy(float* dst, const float* src, long total) {
    long i = gtid_(); if (i < total) dst[i] = src[i];
}

// ------------------- start conv (pad 12->19) + skip0 -----------------------
// x:(B,2,N,12)  h:(B,32,N,19)
__global__ void k_start(const float* __restrict__ x, const float* __restrict__ w,
                        const float* __restrict__ bias, float* __restrict__ h, long total) {
    long i = gtid_(); if (i >= total) return;
    int t = i % 19; long r = i / 19;
    int n = r % N_; r /= N_;
    int c = r % 32; int b = r / 32;
    float acc = bias[c];
    int ts = t - 7;                               // left pad 19-12=7
    if (ts >= 0) {
        acc += w[c * 2 + 0] * x[(((long)b * 2 + 0) * N_ + n) * 12 + ts];
        acc += w[c * 2 + 1] * x[(((long)b * 2 + 1) * N_ + n) * 12 + ts];
    }
    h[i] = acc;
}
// skip0:(B,64,N,1) ; w_sk0:(64,2,1,19)
__global__ void k_skip0(const float* __restrict__ x, const float* __restrict__ w,
                        const float* __restrict__ bias, float* __restrict__ sk, long total) {
    long i = gtid_(); if (i >= total) return;
    int n = i % N_; long r = i / N_;
    int c = r % 64; int b = r / 64;
    float acc = bias[c];
    for (int k = 7; k < 19; ++k) {
        int ts = k - 7;
        acc += w[(c * 2 + 0) * 19 + k] * x[(((long)b * 2 + 0) * N_ + n) * 12 + ts];
        acc += w[(c * 2 + 1) * 19 + k] * x[(((long)b * 2 + 1) * N_ + n) * 12 + ts];
    }
    sk[i] = acc;
}

// ------------------------- GTU concat (k=3,5,7) ----------------------------
// h:(B,32,N,T) -> tc:(B,32,N,3T-12)
__global__ void k_gtu_cat(const float* __restrict__ h,
                          const float* __restrict__ wg3, const float* __restrict__ bg3,
                          const float* __restrict__ wg5, const float* __restrict__ bg5,
                          const float* __restrict__ wg7, const float* __restrict__ bg7,
                          float* __restrict__ tc, int T, long total) {
    long i = gtid_(); if (i >= total) return;
    int L = 3 * T - 12;
    int l = i % L; long r = i / L;
    int n = r % N_; r /= N_;
    int c = r % 32; int b = r / 32;
    int L3 = T - 2, L5 = T - 4;
    const float* wg; const float* bg; int k, t;
    if (l < L3)            { wg = wg3; bg = bg3; k = 3; t = l; }
    else if (l < L3 + L5)  { wg = wg5; bg = bg5; k = 5; t = l - L3; }
    else                   { wg = wg7; bg = bg7; k = 7; t = l - L3 - L5; }
    float yt = bg[c], ys = bg[c + 32];
    for (int ci = 0; ci < 32; ++ci) {
        const float* hp = h + (((long)b * 32 + ci) * N_ + n) * T + t;
        const float* wt = wg + ((long)c * 32 + ci) * k;
        const float* ws = wg + ((long)(c + 32) * 32 + ci) * k;
        for (int j = 0; j < k; ++j) { float hv = hp[j]; yt += hv * wt[j]; ys += hv * ws[j]; }
    }
    tc[i] = tanhf(yt) * sigm_(ys);
}

// ------------- time-FC:  h_out = relu(h_in[...,-Tout:] + tc@W.T + b) -------
__global__ void k_fc_relu_add(const float* __restrict__ tc, const float* __restrict__ W,
                              const float* __restrict__ bias, const float* __restrict__ hin,
                              float* __restrict__ hout, int Th, int L, int Tout, long total) {
    long i = gtid_(); if (i >= total) return;
    int to = i % Tout; long r = i / Tout;
    int n = r % N_; r /= N_;
    int c = r % 32; int b = r / 32;
    float acc = bias[to];
    const float* tp = tc + (((long)b * 32 + c) * N_ + n) * L;
    const float* wp = W + (long)to * L;
    for (int a = 0; a < L; ++a) acc += tp[a] * wp[a];
    float hv = hin[(((long)b * 32 + c) * N_ + n) * Th + (Th - Tout) + to];
    hout[i] = relu_(hv + acc);
}

// --------------------- skip conv (out time always 7) -----------------------
__global__ void k_skipconv(const float* __restrict__ h, const float* __restrict__ w,
                           const float* __restrict__ bias, int Th, int ks,
                           const float* __restrict__ skOld, int TsOld,
                           float* __restrict__ skNew, long total) {
    long i = gtid_(); if (i >= total) return;
    int t = i % 7; long r = i / 7;
    int n = r % N_; r /= N_;
    int c = r % 64; int b = r / 64;
    float acc = bias[c];
    for (int ci = 0; ci < 32; ++ci) {
        const float* hp = h + (((long)b * 32 + ci) * N_ + n) * Th + t;
        const float* wp = w + ((long)c * 32 + ci) * ks;
        for (int j = 0; j < ks; ++j) acc += hp[j] * wp[j];
    }
    int ts = (TsOld == 1) ? 0 : t;
    acc += skOld[(((long)b * 64 + c) * N_ + n) * TsOld + ts];
    skNew[i] = acc;
}

// ------------------ (B,32,N,T) -> xp:(B,T,N,F) transpose -------------------
__global__ void k_to_xp(const float* __restrict__ h, float* __restrict__ xp, int T, long total) {
    long i = gtid_(); if (i >= total) return;   // i indexes xp layout
    int f = i % F_; long r = i / F_;
    int n = r % N_; r /= N_;
    int t = r % T; int b = r / T;
    xp[i] = h[(((long)b * 32 + f) * N_ + n) * T + t];
}

// --------------------------- temporal attention ----------------------------
__global__ void k_tatt_L1(const float* __restrict__ xp, const float* __restrict__ u1,
                          float* __restrict__ L1, int T, long total) {
    long i = gtid_(); if (i >= total) return;   // (B,T,F)
    int f = i % F_; long r = i / F_;
    int t = r % T; int b = r / T;
    float acc = 0.0f;
    const float* xb = xp + (((long)b * T + t) * N_) * F_ + f;
    for (int n = 0; n < N_; ++n) acc += xb[(long)n * F_] * u1[n];
    L1[i] = acc;
}
__global__ void k_tatt_lr(const float* __restrict__ xp, const float* __restrict__ L1,
                          const float* __restrict__ u2, const float* __restrict__ u3,
                          float* __restrict__ lhs, float* __restrict__ rhs, int T, long total) {
    long i = gtid_(); if (i >= total) return;   // (B,T,N)
    int n = i % N_; long r = i / N_;
    int t = r % T; int b = r / T;
    float a = 0.0f, c = 0.0f;
    const float* l1 = L1 + ((long)b * T + t) * F_;
    const float* xb = xp + (((long)b * T + t) * N_ + n) * F_;
    for (int f = 0; f < F_; ++f) { a += l1[f] * u2[(long)f * N_ + n]; c += u3[f] * xb[f]; }
    lhs[((long)b * T + t) * N_ + n] = a;
    rhs[((long)b * N_ + n) * T + t] = c;
}
__global__ void k_tatt_ps(const float* __restrict__ lhs, const float* __restrict__ rhs,
                          const float* __restrict__ be, float* __restrict__ ps, int T, long total) {
    long i = gtid_(); if (i >= total) return;   // (B,T,T)
    int m = i % T; long r = i / T;
    int t = r % T; int b = r / T;
    float acc = 0.0f;
    const float* lp = lhs + ((long)b * T + t) * N_;
    const float* rp = rhs + (long)b * N_ * T + m;
    for (int n = 0; n < N_; ++n) acc += lp[n] * rp[(long)n * T];
    ps[i] = sigm_(acc + be[t * T + m]);
}
__global__ void k_tatt_ve(const float* __restrict__ ps, const float* __restrict__ ve,
                          float* __restrict__ e, int T, long total) {
    long i = gtid_(); if (i >= total) return;   // (B,T,T): e[b,i2,m]
    int m = i % T; long r = i / T;
    int i2 = r % T; int b = r / T;
    float acc = 0.0f;
    for (int t = 0; t < T; ++t) acc += ve[i2 * T + t] * ps[((long)b * T + t) * T + m];
    e[i] = acc;
}
// softmax over axis 1 of (B,T,T): one thread per (b,m)
__global__ void k_softmax_d1(float* __restrict__ e, int T, long total) {
    long i = gtid_(); if (i >= total) return;
    int m = i % T; int b = i / T;
    float* base = e + (long)b * T * T + m;
    float mx = -3.0e38f;
    for (int t = 0; t < T; ++t) { float v = base[(long)t * T]; mx = v > mx ? v : mx; }
    float s = 0.0f;
    for (int t = 0; t < T; ++t) s += expf(base[(long)t * T] - mx);
    float inv = 1.0f / s;
    for (int t = 0; t < T; ++t) base[(long)t * T] = expf(base[(long)t * T] - mx) * inv;
}
// x_tat[b,t,n,f] = sum_t2 e[b,t,t2]*xp[b,t2,n,f]
__global__ void k_xtat(const float* __restrict__ e, const float* __restrict__ xp,
                       float* __restrict__ xt, int T, long total) {
    long i = gtid_(); if (i >= total) return;
    int f = i % F_; long r = i / F_;
    int n = r % N_; r /= N_;
    int t = r % T; int b = r / T;
    float acc = 0.0f;
    const float* ep = e + ((long)b * T + t) * T;
    for (int t2 = 0; t2 < T; ++t2) acc += ep[t2] * xp[(((long)b * T + t2) * N_ + n) * F_ + f];
    xt[i] = acc;
}

// ---------------------------- spatial attention ----------------------------
__global__ void k_satt_L1(const float* __restrict__ xt, const float* __restrict__ w1,
                          float* __restrict__ L1, int T, long total) {
    long i = gtid_(); if (i >= total) return;   // (B,N,F)
    int f = i % F_; long r = i / F_;
    int n = r % N_; int b = r / N_;
    float acc = 0.0f;
    for (int t = 0; t < T; ++t) acc += xt[(((long)b * T + t) * N_ + n) * F_ + f] * w1[t];
    L1[i] = acc;
}
__global__ void k_satt_lr(const float* __restrict__ xt, const float* __restrict__ L1,
                          const float* __restrict__ w2, const float* __restrict__ w3,
                          float* __restrict__ lhs, float* __restrict__ rhs, int T, long total) {
    long i = gtid_(); if (i >= total) return;   // (B,N,T)
    int t = i % T; long r = i / T;
    int n = r % N_; int b = r / N_;
    float a = 0.0f, c = 0.0f;
    const float* l1 = L1 + ((long)b * N_ + n) * F_;
    const float* xb = xt + (((long)b * T + t) * N_ + n) * F_;
    for (int f = 0; f < F_; ++f) { a += l1[f] * w2[f * T + t]; c += w3[f] * xb[f]; }
    lhs[((long)b * N_ + n) * T + t] = a;
    rhs[((long)b * T + t) * N_ + n] = c;
}
// sigT[b,m,n] = bf16(sigmoid(lhs[b,n,:].rhs[b,:,m] + bs[n,m]))  (B^T, Kp-padded)
__global__ void k_satt_sigT(const float* __restrict__ lhs, const float* __restrict__ rhs,
                            const float* __restrict__ bs, __bf16* __restrict__ sigT,
                            int T, long total) {
    long i = gtid_(); if (i >= total) return;   // (B, N(m), KP(n))
    int n = i % KP_; long r = i / KP_;
    int m = r % N_; int b = r / N_;
    if (n >= N_) { sigT[i] = (__bf16)0.0f; return; }
    float acc = 0.0f;
    const float* lp = lhs + ((long)b * N_ + n) * T;
    const float* rp = rhs + (long)b * T * N_ + m;
    for (int t = 0; t < T; ++t) acc += lp[t] * rp[(long)t * N_];
    sigT[i] = (__bf16)sigm_(acc + bs[(long)n * N_ + m]);
}
// vsp[row, k] = bf16(vs[row,k]) with K zero-padded to KP
__global__ void k_pack_vs(const float* __restrict__ vs, __bf16* __restrict__ vsp, long total) {
    long i = gtid_(); if (i >= total) return;   // (N, KP)
    int k = i % KP_; int row = i / KP_;
    vsp[i] = (k < N_) ? (__bf16)vs[(long)row * N_ + k] : (__bf16)0.0f;
}
// softmax over axis 1 of (B,N,N): one thread per (b,m)
__global__ void k_softmax_col(float* __restrict__ s, long total) {
    long i = gtid_(); if (i >= total) return;
    int m = i % N_; int b = i / N_;
    float* base = s + (long)b * N_ * N_ + m;
    float mx = -3.0e38f;
    for (int n = 0; n < N_; ++n) { float v = base[(long)n * N_]; mx = v > mx ? v : mx; }
    float sum = 0.0f;
    for (int n = 0; n < N_; ++n) sum += expf(base[(long)n * N_] - mx);
    float inv = 1.0f / sum;
    for (int n = 0; n < N_; ++n) base[(long)n * N_] = expf(base[(long)n * N_] - mx) * inv;
}

// ------------------- batched bf16 GEMM via v_wmma (fp32 acc) ---------------
// C[M,Nn] = A[M,Kp] (row-major bf16, zero-padded K) x Bt[Nn,Kp] (B transposed,
// row-major bf16, zero-padded K). grid.z = batch. 4 waves/block; each wave
// owns a 32x32 C tile (2x2 WMMA accumulators). Per K-step: 8x b128 loads feed
// 4x v_wmma_f32_16x16x32_bf16 -- no bounds checks inside the loop (K padded;
// M/Nn edges handled by row/col clamping, which is safe because an A row only
// contributes to the same C row and a B column only to the same C column).
__global__ __launch_bounds__(128)
void k_gemm_bf16(const __bf16* __restrict__ A, const __bf16* __restrict__ Bt,
                 float* __restrict__ C, int M, int Nn, int Kp,
                 long sA, long sB, long sC) {
    const __bf16* Ab = A  + (long)blockIdx.z * sA;
    const __bf16* Bb = Bt + (long)blockIdx.z * sB;
    float* Cb = C + (long)blockIdx.z * sC;
    const int lane = threadIdx.x & 31;
    const int wave = threadIdx.x >> 5;
    const int half = lane >> 4;
    const int r15  = lane & 15;
    const int mBase = blockIdx.x * 64 + (wave >> 1) * 32;
    const int nBase = blockIdx.y * 64 + (wave & 1) * 32;

    const int row0 = mBase + r15, row1 = row0 + 16;
    const int col0 = nBase + r15, col1 = col0 + 16;
    const int row0c = row0 < M ? row0 : M - 1;
    const int row1c = row1 < M ? row1 : M - 1;
    const int col0c = col0 < Nn ? col0 : Nn - 1;
    const int col1c = col1 < Nn ? col1 : Nn - 1;
    const __bf16* a0p = Ab + (long)row0c * Kp + half * 8;
    const __bf16* a1p = Ab + (long)row1c * Kp + half * 8;
    const __bf16* b0p = Bb + (long)col0c * Kp + half * 8;
    const __bf16* b1p = Bb + (long)col1c * Kp + half * 8;

    v8f acc00 = (v8f){0.f,0.f,0.f,0.f,0.f,0.f,0.f,0.f};
    v8f acc01 = acc00, acc10 = acc00, acc11 = acc00;

    for (int kb = 0; kb < Kp; kb += 32) {
        v8bf aL0 = *(const v8bf*)(a0p + kb);
        v8bf aH0 = *(const v8bf*)(a0p + kb + 16);
        v8bf aL1 = *(const v8bf*)(a1p + kb);
        v8bf aH1 = *(const v8bf*)(a1p + kb + 16);
        v8bf bL0 = *(const v8bf*)(b0p + kb);
        v8bf bH0 = *(const v8bf*)(b0p + kb + 16);
        v8bf bL1 = *(const v8bf*)(b1p + kb);
        v8bf bH1 = *(const v8bf*)(b1p + kb + 16);
        v16bf a0 = __builtin_shufflevector(aL0, aH0, 0,1,2,3,4,5,6,7,8,9,10,11,12,13,14,15);
        v16bf a1 = __builtin_shufflevector(aL1, aH1, 0,1,2,3,4,5,6,7,8,9,10,11,12,13,14,15);
        v16bf b0 = __builtin_shufflevector(bL0, bH0, 0,1,2,3,4,5,6,7,8,9,10,11,12,13,14,15);
        v16bf b1 = __builtin_shufflevector(bL1, bH1, 0,1,2,3,4,5,6,7,8,9,10,11,12,13,14,15);
        acc00 = __builtin_amdgcn_wmma_f32_16x16x32_bf16(false, a0, false, b0, (short)0, acc00, false, false);
        acc01 = __builtin_amdgcn_wmma_f32_16x16x32_bf16(false, a0, false, b1, (short)0, acc01, false, false);
        acc10 = __builtin_amdgcn_wmma_f32_16x16x32_bf16(false, a1, false, b0, (short)0, acc10, false, false);
        acc11 = __builtin_amdgcn_wmma_f32_16x16x32_bf16(false, a1, false, b1, (short)0, acc11, false, false);
    }
#pragma unroll
    for (int j = 0; j < 8; ++j) {
        const int r0 = mBase + j + half * 8;
        const int r1 = r0 + 16;
        const int c0 = nBase + r15;
        const int c1 = c0 + 16;
        if (r0 < M && c0 < Nn) Cb[(long)r0 * Nn + c0] = acc00[j];
        if (r0 < M && c1 < Nn) Cb[(long)r0 * Nn + c1] = acc01[j];
        if (r1 < M && c0 < Nn) Cb[(long)r1 * Nn + c0] = acc10[j];
        if (r1 < M && c1 < Nn) Cb[(long)r1 * Nn + c1] = acc11[j];
    }
}

// ------------------------------ Chebyshev ----------------------------------
// At[k,b,n,m] = bf16(cheb[k,m,n] * s[b,m,n])   (A^T layout, Kp-padded over m)
__global__ void k_cheb_A(const float* __restrict__ cheb, const float* __restrict__ s,
                         __bf16* __restrict__ At, long total) {
    long i = gtid_(); if (i >= total) return;   // (3,B,N(n),KP(m))
    int m = i % KP_; long r = i / KP_;
    int n = r % N_; r /= N_;
    int b = r % B_; int k = r / B_;
    At[i] = (m < N_)
        ? (__bf16)(cheb[(long)k * N_ * N_ + (long)m * N_ + n] * s[((long)b * N_ + m) * N_ + n])
        : (__bf16)0.0f;
}
// XrT[b, t*F+f, m] = bf16(xp[b,t,m,f])   (B transposed for GEMM, Kp-padded)
__global__ void k_xrT(const float* __restrict__ xp, __bf16* __restrict__ XrT, int T, long total) {
    long i = gtid_(); if (i >= total) return;   // (B, T*F, KP(m))
    int m = i % KP_; long r = i / KP_;
    int tf = r % (T * F_); int b = r / (T * F_);
    if (m >= N_) { XrT[i] = (__bf16)0.0f; return; }
    int t = tf / F_, f = tf % F_;
    XrT[i] = (__bf16)xp[(((long)b * T + t) * N_ + m) * F_ + f];
}
// hacc[b,f,n,t] += sum_fp tmp[b,n,t*F+fp] * theta[fp,f] ; optional final relu
__global__ void k_theta_acc(const float* __restrict__ tmp, const float* __restrict__ theta,
                            float* __restrict__ hacc, int T, int doRelu, long total) {
    long i = gtid_(); if (i >= total) return;
    int t = i % T; long r = i / T;
    int n = r % N_; r /= N_;
    int f = r % F_; int b = r / F_;
    float acc = hacc[i];
    const float* tp = tmp + ((long)b * N_ + n) * (T * F_) + t * F_;
    for (int fp = 0; fp < F_; ++fp) acc += tp[fp] * theta[fp * F_ + f];
    hacc[i] = doRelu ? relu_(acc) : acc;
}

// --------------------- residual add + custom layernorm ---------------------
__global__ void k_res_add(float* __restrict__ h, const float* __restrict__ res,
                          int T, int Tres, long total) {
    long i = gtid_(); if (i >= total) return;
    int t = i % T; long r = i / T;
    int n = r % N_; r /= N_;
    int c = r % 32; int b = r / 32;
    h[i] += res[(((long)b * 32 + c) * N_ + n) * Tres + (Tres - T) + t];
}
__global__ void k_ln_stats(const float* __restrict__ h, float* __restrict__ stats, long cnt) {
    __shared__ float ss[256], sq[256];
    int b = blockIdx.x; int tid = threadIdx.x;
    const float* base = h + (long)b * cnt;
    float s = 0.0f, q = 0.0f;
    for (long i = tid; i < cnt; i += 256) { float v = base[i]; s += v; q += v * v; }
    ss[tid] = s; sq[tid] = q; __syncthreads();
    for (int o = 128; o > 0; o >>= 1) {
        if (tid < o) { ss[tid] += ss[tid + o]; sq[tid] += sq[tid + o]; }
        __syncthreads();
    }
    if (tid == 0) {
        float m = ss[0] / (float)cnt;
        stats[2 * b] = m;
        stats[2 * b + 1] = sq[0] / (float)cnt - m * m;
    }
}
__global__ void k_ln_apply(float* __restrict__ h, const float* __restrict__ stats,
                           const float* __restrict__ wn, const float* __restrict__ bn,
                           int T, long total) {
    long i = gtid_(); if (i >= total) return;
    int t = i % T; long r = i / T;
    int n = r % N_; r /= N_;
    int c = r % 32; int b = r / 32;
    float m = stats[2 * b], v = stats[2 * b + 1];
    float inv = rsqrtf(v + 1e-5f);
    long wi = ((long)c * N_ + n) * T + t;
    h[i] = (h[i] - m) * inv * wn[wi] + bn[wi];
}

// ------------------------------- output head -------------------------------
__global__ void k_final1(const float* __restrict__ h, const float* __restrict__ w,
                         const float* __restrict__ bias, const float* __restrict__ sk,
                         float* __restrict__ skf, long total) {
    long i = gtid_(); if (i >= total) return;   // (B,64,N)
    int n = i % N_; long r = i / N_;
    int c = r % 64; int b = r / 64;
    float acc = bias[c] + sk[(((long)b * 64 + c) * N_ + n) * 7 + 6];
    for (int ci = 0; ci < 32; ++ci)
        acc += w[c * 32 + ci] * h[(((long)b * 32 + ci) * N_ + n)];
    skf[i] = relu_(acc);
}
__global__ void k_final2(const float* __restrict__ skf, const float* __restrict__ w,
                         const float* __restrict__ bias, float* __restrict__ m1, long total) {
    long i = gtid_(); if (i >= total) return;   // (B,128,N)
    int n = i % N_; long r = i / N_;
    int o = r % 128; int b = r / 128;
    float acc = bias[o];
    for (int c = 0; c < 64; ++c) acc += w[o * 64 + c] * skf[((long)b * 64 + c) * N_ + n];
    m1[i] = relu_(acc);
}
__global__ void k_final3(const float* __restrict__ m1, const float* __restrict__ w,
                         const float* __restrict__ bias, float* __restrict__ out, long total) {
    long i = gtid_(); if (i >= total) return;   // (B,12,N)
    int n = i % N_; long r = i / N_;
    int o = r % 12; int b = r / 12;
    float acc = bias[o];
    for (int c = 0; c < 128; ++c) acc += w[o * 128 + c] * m1[((long)b * 128 + c) * N_ + n];
    out[i] = acc;
}

// ---------------------------------------------------------------------------
#define LAUNCH(kern, total, ...)                                                   \
    kern<<<dim3((unsigned)(((total) + 255) / 256)), dim3(256), 0, stream>>>(__VA_ARGS__)

extern "C" void kernel_launch(void* const* d_in, const int* in_sizes, int n_in,
                              void* d_out, int out_size, void* d_ws, size_t ws_size,
                              hipStream_t stream) {
    (void)in_sizes; (void)n_in; (void)out_size; (void)ws_size;
    auto IN = [&](int i) { return (const float*)d_in[i]; };
    const float* x       = IN(0);
    const float* w_start = IN(1);  const float* b_start = IN(2);
    const float* w_sk0   = IN(3);  const float* b_sk0   = IN(4);
    const float* wg3 = IN(5),  *bg3 = IN(6);
    const float* wg5 = IN(7),  *bg5 = IN(8);
    const float* wg7 = IN(9),  *bg7 = IN(10);
    const float* cheb  = IN(59);
    const float* theta = IN(60);
    const float* w_se = IN(67), *b_se = IN(68);
    const float* we1  = IN(69), *be1  = IN(70);
    const float* we2  = IN(71), *be2  = IN(72);

    // ---- carve workspace ----
    char* wsb = (char*)d_ws;
    size_t off = 0;
    auto carve = [&](size_t bytes) -> void* {
        void* p = wsb + off;
        off = (off + bytes + 255) & ~(size_t)255;
        return p;
    };
    const long HMAX = (long)B_ * 32 * N_ * 19;          // 9,728,000
    float* h    = (float*)carve(HMAX * 4);
    float* h2   = (float*)carve(HMAX * 4);
    float* res  = (float*)carve(HMAX * 4);
    float* xp   = (float*)carve(HMAX * 4);
    float* xt   = (float*)carve(HMAX * 4);
    float* tc   = (float*)carve((long)B_ * 32 * N_ * 45 * 4);
    float* skA  = (float*)carve((long)B_ * 64 * N_ * 7 * 4);
    float* skB  = (float*)carve((long)B_ * 64 * N_ * 7 * 4);
    float* L1t  = (float*)carve((long)B_ * 19 * F_ * 4);
    float* lhsT = (float*)carve((long)B_ * 19 * N_ * 4);
    float* rhsT = (float*)carve((long)B_ * N_ * 19 * 4);
    float* psT  = (float*)carve((long)B_ * 19 * 19 * 4);
    float* ebuf = (float*)carve((long)B_ * 19 * 19 * 4);
    float* L1s  = (float*)carve((long)B_ * N_ * F_ * 4);
    float* lhsS = (float*)carve((long)B_ * N_ * 19 * 4);
    float* rhsS = (float*)carve((long)B_ * 19 * N_ * 4);
    __bf16* sigT = (__bf16*)carve((long)B_ * N_ * KP_ * 2);
    __bf16* vsp  = (__bf16*)carve((long)N_ * KP_ * 2);
    float* sbuf = (float*)carve((long)B_ * N_ * N_ * 4);
    __bf16* At  = (__bf16*)carve((long)3 * B_ * N_ * KP_ * 2);
    __bf16* XrT = (__bf16*)carve((long)B_ * 19 * F_ * KP_ * 2);
    float* tmpC = (float*)carve((long)B_ * N_ * 19 * F_ * 4);
    float* stats = (float*)carve((long)B_ * 2 * 4);
    float* skf  = (float*)carve((long)B_ * 64 * N_ * 4);
    float* m1b  = (float*)carve((long)B_ * 128 * N_ * 4);

    // ---- stem ----
    LAUNCH(k_start, (long)B_ * 32 * N_ * 19, x, w_start, b_start, h, (long)B_ * 32 * N_ * 19);
    LAUNCH(k_skip0, (long)B_ * 64 * N_, x, w_sk0, b_sk0, skA, (long)B_ * 64 * N_);
    int TsOld = 1;

    const int TinA[3]  = {19, 13, 7};
    const int ToutA[3] = {13, 7, 1};
    const int kscA[3]  = {13, 7, 1};

    for (int i = 0; i < 3; ++i) {
        const int T = TinA[i], T2 = ToutA[i];
        const int L = 3 * T - 12, TFv = T * F_;
        const float* wf1 = IN(11 + 2 * i); const float* bf1 = IN(12 + 2 * i);
        const float* wf2 = IN(17 + 2 * i); const float* bf2 = IN(18 + 2 * i);
        const float* wsc = IN(23 + 2 * i); const float* bsc = IN(24 + 2 * i);
        const int ab = 29 + 10 * i;
        const float* u1 = IN(ab + 0); const float* u2 = IN(ab + 1); const float* u3 = IN(ab + 2);
        const float* be = IN(ab + 3); const float* ve = IN(ab + 4);
        const float* w1 = IN(ab + 5); const float* w2 = IN(ab + 6); const float* w3 = IN(ab + 7);
        const float* bs = IN(ab + 8); const float* vs = IN(ab + 9);
        const float* wn = IN(61 + 2 * i); const float* bn = IN(62 + 2 * i);

        const long nH  = (long)B_ * 32 * N_ * T;
        const long nH2 = (long)B_ * 32 * N_ * T2;

        // residual
        LAUNCH(k_copy, nH, res, h, nH);
        // GTU block 1 + fc1 (T -> T)
        LAUNCH(k_gtu_cat, (long)B_ * 32 * N_ * L, h, wg3, bg3, wg5, bg5, wg7, bg7, tc, T,
               (long)B_ * 32 * N_ * L);
        LAUNCH(k_fc_relu_add, nH, tc, wf1, bf1, h, h2, T, L, T, nH);
        { float* t_ = h; h = h2; h2 = t_; }
        // skip conv (out time 7)
        LAUNCH(k_skipconv, (long)B_ * 64 * N_ * 7, h, wsc, bsc, T, kscA[i], skA, TsOld, skB,
               (long)B_ * 64 * N_ * 7);
        { float* t_ = skA; skA = skB; skB = t_; }
        TsOld = 7;
        // xp = (B,T,N,F)
        LAUNCH(k_to_xp, nH, h, xp, T, nH);
        // temporal attention
        LAUNCH(k_tatt_L1, (long)B_ * T * F_, xp, u1, L1t, T, (long)B_ * T * F_);
        LAUNCH(k_tatt_lr, (long)B_ * T * N_, xp, L1t, u2, u3, lhsT, rhsT, T, (long)B_ * T * N_);
        LAUNCH(k_tatt_ps, (long)B_ * T * T, lhsT, rhsT, be, psT, T, (long)B_ * T * T);
        LAUNCH(k_tatt_ve, (long)B_ * T * T, psT, ve, ebuf, T, (long)B_ * T * T);
        LAUNCH(k_softmax_d1, (long)B_ * T, ebuf, T, (long)B_ * T);
        LAUNCH(k_xtat, nH, ebuf, xp, xt, T, nH);
        // spatial attention
        LAUNCH(k_satt_L1, (long)B_ * N_ * F_, xt, w1, L1s, T, (long)B_ * N_ * F_);
        LAUNCH(k_satt_lr, (long)B_ * N_ * T, xt, L1s, w2, w3, lhsS, rhsS, T, (long)B_ * N_ * T);
        LAUNCH(k_satt_sigT, (long)B_ * N_ * KP_, lhsS, rhsS, bs, sigT, T, (long)B_ * N_ * KP_);
        LAUNCH(k_pack_vs, (long)N_ * KP_, vs, vsp, (long)N_ * KP_);
        {   // s_pre[b] = vs (500x500) x sig[b] (500x500), fp32 out
            dim3 gg((N_ + 63) / 64, (N_ + 63) / 64, B_);
            k_gemm_bf16<<<gg, dim3(128), 0, stream>>>(vsp, sigT, sbuf, N_, N_, KP_,
                                                      0, (long)N_ * KP_, (long)N_ * N_);
        }
        LAUNCH(k_softmax_col, (long)B_ * N_, sbuf, (long)B_ * N_);
        // Chebyshev graph conv via WMMA
        LAUNCH(k_cheb_A, (long)3 * B_ * N_ * KP_, cheb, sbuf, At, (long)3 * B_ * N_ * KP_);
        LAUNCH(k_xrT, (long)B_ * TFv * KP_, xp, XrT, T, (long)B_ * TFv * KP_);
        LAUNCH(k_zero, nH, h2, nH);
        for (int k = 0; k < 3; ++k) {
            dim3 gg((N_ + 63) / 64, (TFv + 63) / 64, B_);
            k_gemm_bf16<<<gg, dim3(128), 0, stream>>>(At + (long)k * B_ * N_ * KP_, XrT, tmpC,
                                                      N_, TFv, KP_,
                                                      (long)N_ * KP_, (long)TFv * KP_,
                                                      (long)N_ * TFv);
            LAUNCH(k_theta_acc, nH, tmpC, theta + (long)k * F_ * F_, h2, T, (k == 2) ? 1 : 0, nH);
        }
        { float* t_ = h; h = h2; h2 = t_; }
        // GTU block 2 + fc2 (T -> T2)
        LAUNCH(k_gtu_cat, (long)B_ * 32 * N_ * L, h, wg3, bg3, wg5, bg5, wg7, bg7, tc, T,
               (long)B_ * 32 * N_ * L);
        LAUNCH(k_fc_relu_add, nH2, tc, wf2, bf2, h, h2, T, L, T2, nH2);
        { float* t_ = h; h = h2; h2 = t_; }
        // residual + layernorm
        LAUNCH(k_res_add, nH2, h, res, T2, T, nH2);
        k_ln_stats<<<dim3(B_), dim3(256), 0, stream>>>(h, stats, (long)32 * N_ * T2);
        LAUNCH(k_ln_apply, nH2, h, stats, wn, bn, T2, nH2);
    }

    // ---- head ----
    float* out = (float*)d_out;
    LAUNCH(k_final1, (long)B_ * 64 * N_, h, w_se, b_se, skA, skf, (long)B_ * 64 * N_);
    LAUNCH(k_final2, (long)B_ * 128 * N_, skf, we1, be1, m1b, (long)B_ * 128 * N_);
    LAUNCH(k_final3, (long)B_ * 12 * N_, m1b, we2, be2, out, (long)B_ * 12 * N_);
}
